// GNNModel_31464930410955
// MI455X (gfx1250) — matrix operand order repaired and checked by reading, not compiled, not run on hardware
//
#include <hip/hip_runtime.h>
#include <hip/hip_bf16.h>

// ---------------- problem constants (match reference) ----------------
#define NNODES 50000
#define NEDGES 500000
#define DIN    256
#define HID    64
#define HEADS  4
#define HC     (HEADS * HID)   // 256
#define NEG_SLOPE 0.2f

// ---------------- types for WMMA fragments ----------------
typedef __bf16 bf16x8  __attribute__((ext_vector_type(8)));
typedef __bf16 bf16x16 __attribute__((ext_vector_type(16)));
typedef float  f32x8   __attribute__((ext_vector_type(8)));

union BFrag { bf16x16 v; bf16x8 p[2]; };

// order-preserving float<->uint key for atomicMax-based segment max
__device__ __forceinline__ unsigned enc_key(float f) {
  unsigned u = __float_as_uint(f);
  return (u & 0x80000000u) ? ~u : (u | 0x80000000u);
}
__device__ __forceinline__ float dec_key(unsigned k) {
  unsigned u = (k & 0x80000000u) ? (k & 0x7FFFFFFFu) : ~k;
  return __uint_as_float(u);
}
// host-side: key of -inf (sign bit set -> ~bits) == 0x007FFFFF
static inline unsigned enc_host_minus_inf() { return ~0xFF800000u; }

// ---------------- small utility kernels ----------------
__global__ void fill_f32_k(float* p, float v, long n) {
  long i = (long)blockIdx.x * blockDim.x + threadIdx.x;
  if (i < n) p[i] = v;
}
__global__ void fill_u32_k(unsigned* p, unsigned v, long n) {
  long i = (long)blockIdx.x * blockDim.x + threadIdx.x;
  if (i < n) p[i] = v;
}
__global__ void cvt_bf16_k(const float* __restrict__ x, __bf16* __restrict__ y, long n) {
  long i = (long)blockIdx.x * blockDim.x + threadIdx.x;
  if (i < n) y[i] = (__bf16)x[i];
}
// Wt[n*rows + k] = W[k*cols + n]  (rows = K dim, cols = N dim)
__global__ void transpose_cvt_k(const float* __restrict__ W, __bf16* __restrict__ Wt,
                                int rows, int cols) {
  long t = (long)blockIdx.x * blockDim.x + threadIdx.x;
  if (t >= (long)rows * cols) return;
  int n = (int)(t / rows);
  int k = (int)(t % rows);
  Wt[(size_t)n * rows + k] = (__bf16)W[(size_t)k * cols + n];
}
// y_bf16 = bf16(relu(acc + bias[col]))
__global__ void bias_relu_cvt_k(const float* __restrict__ acc, const float* __restrict__ bias,
                                __bf16* __restrict__ y, long n, int cols) {
  long i = (long)blockIdx.x * blockDim.x + threadIdx.x;
  if (i >= n) return;
  float v = acc[i] + bias[i % cols];
  y[i] = (__bf16)(v > 0.f ? v : 0.f);
}

// ---------------- bf16 WMMA GEMM: C[M,N] = A[M,K] * Bt[N,K]^T ----------------
// One wave (32 threads, one wave per block) computes a 16(M) x 64(N) tile:
// 4 accumulators, A fragment loaded once per K-step and reused by 4 WMMAs.
// K stepped by 32 (v_wmma_f32_16x16x32_bf16). A/B k order is contiguous (16*half+s), not ISA table; see
// CDNA5 ISA 7.12.2 (16-bit A 16x32, B 32x16, f32 C/D 16x16).
// Requires N % 64 == 0, M % 16 == 0, K % 32 == 0 (all exact here).
__global__ void __launch_bounds__(32)
gemm_bf16_wmma_k(const __bf16* __restrict__ A, const __bf16* __restrict__ Bt,
                 float* __restrict__ C, int M, int N, int K) {
  int wave = blockIdx.x;                    // one wave per block
  int lane = threadIdx.x & 31;
  int tilesN = N >> 6;                      // 64-wide column tiles
  int tm = wave / tilesN;
  int tn = wave - tm * tilesN;

  int hl  = lane >> 4;                      // half-wave select
  int l16 = lane & 15;

  // A: row m = tm*16 + l16; lanes 0-15 carry K 0..15, lanes 16-31 carry K 16..31
  const __bf16* arow = A + (size_t)(tm * 16 + l16) * K + hl * 16;
  // B (Bt rows are columns of B): lane col = base + l16; lanes 0-15 K 0..15, lanes 16-31 K 16..31
  const __bf16* brow[4];
#pragma unroll
  for (int j = 0; j < 4; ++j)
    brow[j] = Bt + (size_t)(tn * 64 + j * 16 + l16) * K + hl * 16;

  f32x8 acc[4];
#pragma unroll
  for (int j = 0; j < 4; ++j) acc[j] = (f32x8){0.f, 0.f, 0.f, 0.f, 0.f, 0.f, 0.f, 0.f};

  for (int k0 = 0; k0 < K; k0 += 32) {
    BFrag a;
    a.p[0] = *(const bf16x8*)(arow + k0);
    a.p[1] = *(const bf16x8*)(arow + k0 + 8);
    BFrag b[4];
#pragma unroll
    for (int j = 0; j < 4; ++j) {
      b[j].p[0] = *(const bf16x8*)(brow[j] + k0);
      b[j].p[1] = *(const bf16x8*)(brow[j] + k0 + 8);
    }
    if (k0 + 32 < K) {                      // global_prefetch_b8 of next K-slab
      __builtin_prefetch(arow + k0 + 32, 0, 3);
      __builtin_prefetch(brow[0] + k0 + 32, 0, 3);
      __builtin_prefetch(brow[2] + k0 + 32, 0, 3);
    }
#pragma unroll
    for (int j = 0; j < 4; ++j)
      acc[j] = __builtin_amdgcn_wmma_f32_16x16x32_bf16(false, a.v, false, b[j].v,
                                                       (short)0, acc[j], false, false);
  }

  // f32 C/D: VGPR v holds row (hl*8 + v), col (strip base + l16)
  float* cbase = C + (size_t)(tm * 16 + hl * 8) * N + tn * 64 + l16;
#pragma unroll
  for (int j = 0; j < 4; ++j) {
#pragma unroll
    for (int v = 0; v < 8; ++v)
      cbase[(size_t)v * N + j * 16] = acc[j][v];
  }
}

// ---------------- GAT attention coefficients per (node, head) ----------------
__global__ void attn_coeff_k(const float* __restrict__ hfeat,
                             const float* __restrict__ a_src, const float* __restrict__ a_dst,
                             float* __restrict__ asrc, float* __restrict__ adst, int Nn) {
  int t = blockIdx.x * blockDim.x + threadIdx.x;
  if (t >= Nn * HEADS) return;
  int n = t >> 2, hh = t & 3;
  const float* hp = hfeat + (size_t)n * HC + hh * HID;
  const float* sp = a_src + hh * HID;
  const float* dp = a_dst + hh * HID;
  float vs = 0.f, vd = 0.f;
#pragma unroll 8
  for (int c = 0; c < HID; ++c) { float xv = hp[c]; vs += xv * sp[c]; vd += xv * dp[c]; }
  asrc[t] = vs;
  adst[t] = vd;
}

// ---------------- edge phase 1: segment max of logits + in-degree ----------------
__global__ void gat_edge_max_deg_k(const long long* __restrict__ ei, int E, int Nn,
                                   const float* __restrict__ asrc, const float* __restrict__ adst,
                                   unsigned* __restrict__ maxb, float* __restrict__ deg) {
  int e = blockIdx.x * blockDim.x + threadIdx.x;
  if (e >= E + Nn) return;
  long long s, d;
  if (e < E) { s = ei[e]; d = ei[E + e]; } else { s = d = e - E; }
#pragma unroll
  for (int hh = 0; hh < HEADS; ++hh) {
    float v = asrc[s * HEADS + hh] + adst[d * HEADS + hh];
    v = v > 0.f ? v : NEG_SLOPE * v;
    atomicMax(&maxb[d * HEADS + hh], enc_key(v));
  }
  atomicAdd(&deg[d], 1.0f);
}

// ---------------- edge phase 2: sum of exp(logit - max) ----------------
__global__ void gat_edge_expsum_k(const long long* __restrict__ ei, int E, int Nn,
                                  const float* __restrict__ asrc, const float* __restrict__ adst,
                                  const unsigned* __restrict__ maxb, float* __restrict__ sumb) {
  int e = blockIdx.x * blockDim.x + threadIdx.x;
  if (e >= E + Nn) return;
  long long s, d;
  if (e < E) { s = ei[e]; d = ei[E + e]; } else { s = d = e - E; }
#pragma unroll
  for (int hh = 0; hh < HEADS; ++hh) {
    float v = asrc[s * HEADS + hh] + adst[d * HEADS + hh];
    v = v > 0.f ? v : NEG_SLOPE * v;
    float m = dec_key(maxb[d * HEADS + hh]);
    atomicAdd(&sumb[d * HEADS + hh], __expf(v - m));
  }
}

// ---------------- edge phase 3: alpha-weighted message aggregation ----------------
// one wave per (edge, head): 32 lanes x float2 == 64 channels, coalesced 256B/wave
__global__ void gat_edge_message_k(const long long* __restrict__ ei, int E, int Nn,
                                   const float* __restrict__ hfeat,
                                   const float* __restrict__ asrc, const float* __restrict__ adst,
                                   const unsigned* __restrict__ maxb, const float* __restrict__ sumb,
                                   float* __restrict__ accum) {
  int gw = (int)(((long)blockIdx.x * blockDim.x + threadIdx.x) >> 5);
  int lane = threadIdx.x & 31;
  int e = gw >> 2;                          // HEADS == 4
  int hh = gw & 3;
  if (e >= E + Nn) return;
  long long s, d;
  if (e < E) { s = ei[e]; d = ei[E + e]; } else { s = d = e - E; }
  float v = asrc[s * HEADS + hh] + adst[d * HEADS + hh];
  v = v > 0.f ? v : NEG_SLOPE * v;
  float m = dec_key(maxb[d * HEADS + hh]);
  float alpha = __expf(v - m) / (sumb[d * HEADS + hh] + 1e-16f);
  int c = lane * 2;
  const float2 hv = *(const float2*)(hfeat + (size_t)s * HC + hh * HID + c);
  float* dp = accum + (size_t)d * HC + hh * HID + c;
  atomicAdd(dp,     hv.x * alpha);
  atomicAdd(dp + 1, hv.y * alpha);
}

// ---------------- GCN normalized aggregation: one wave per edge ----------------
__global__ void gcn_edge_k(const long long* __restrict__ ei, int E, int Nn,
                           const float* __restrict__ h2, const float* __restrict__ deg,
                           float* __restrict__ accum) {
  int gw = (int)(((long)blockIdx.x * blockDim.x + threadIdx.x) >> 5);
  int lane = threadIdx.x & 31;
  if (gw >= E + Nn) return;
  long long s, d;
  if (gw < E) { s = ei[gw]; d = ei[E + gw]; } else { s = d = gw - E; }
  float ds = deg[s], dd = deg[d];
  float norm = (ds > 0.f ? rsqrtf(ds) : 0.f) * (dd > 0.f ? rsqrtf(dd) : 0.f);
  int c = lane * 2;
  const float2 hv = *(const float2*)(h2 + (size_t)s * HID + c);
  float* dp = accum + (size_t)d * HID + c;
  atomicAdd(dp,     hv.x * norm);
  atomicAdd(dp + 1, hv.y * norm);
}

// ---------------- fused MLP head: relu(x2) -> relu(x2@W1+b1) -> @W2+b2 -> sigmoid ----------------
__global__ void mlp_head_k(const float* __restrict__ gcnacc, const float* __restrict__ b_gcn,
                           const float* __restrict__ W1, const float* __restrict__ b1,
                           const float* __restrict__ W2, const float* __restrict__ b2,
                           float* __restrict__ out, int Nn) {
  int n = blockIdx.x * blockDim.x + threadIdx.x;
  if (n >= Nn) return;
  float x2[HID];
#pragma unroll 8
  for (int c = 0; c < HID; ++c) {
    float v = gcnacc[(size_t)n * HID + c] + b_gcn[c];
    x2[c] = v > 0.f ? v : 0.f;
  }
  float z = b2[0];
  for (int j = 0; j < HID / 2; ++j) {
    float s = b1[j];
#pragma unroll 8
    for (int c = 0; c < HID; ++c) s += x2[c] * W1[c * (HID / 2) + j];
    s = s > 0.f ? s : 0.f;
    z += s * W2[j];
  }
  out[n] = 1.f / (1.f + __expf(-z));
}

// ---------------- host-side orchestration ----------------
extern "C" void kernel_launch(void* const* d_in, const int* in_sizes, int n_in,
                              void* d_out, int out_size, void* d_ws, size_t ws_size,
                              hipStream_t stream) {
  const float*     x     = (const float*)d_in[0];
  const long long* ei    = (const long long*)d_in[1];   // int64 [2,E] flat
  const float*     W_gat = (const float*)d_in[2];
  const float*     a_src = (const float*)d_in[3];
  const float*     a_dst = (const float*)d_in[4];
  const float*     b_gat = (const float*)d_in[5];
  const float*     W_gcn = (const float*)d_in[6];
  const float*     b_gcn = (const float*)d_in[7];
  const float*     W1    = (const float*)d_in[8];
  const float*     b1    = (const float*)d_in[9];
  const float*     W2    = (const float*)d_in[10];
  const float*     b2    = (const float*)d_in[11];
  float* out = (float*)d_out;

  char* wp = (char*)d_ws;
  auto carve = [&](size_t bytes) -> void* {
    void* p = (void*)wp;
    wp += (bytes + 255) & ~(size_t)255;
    return p;
  };
  __bf16*   x_bf   = (__bf16*)  carve((size_t)NNODES * HC * 2);
  __bf16*   WgatT  = (__bf16*)  carve((size_t)DIN * HC * 2);
  float*    hfeat  = (float*)   carve((size_t)NNODES * HC * 4);
  float*    asrc   = (float*)   carve((size_t)NNODES * HEADS * 4);
  float*    adst   = (float*)   carve((size_t)NNODES * HEADS * 4);
  unsigned* maxb   = (unsigned*)carve((size_t)NNODES * HEADS * 4);
  float*    sumb   = (float*)   carve((size_t)NNODES * HEADS * 4);
  float*    deg    = (float*)   carve((size_t)NNODES * 4);
  float*    gatacc = (float*)   carve((size_t)NNODES * HC * 4);
  __bf16*   x1_bf  = (__bf16*)  carve((size_t)NNODES * HC * 2);
  __bf16*   WgcnT  = (__bf16*)  carve((size_t)HID * HC * 2);
  float*    h2     = (float*)   carve((size_t)NNODES * HID * 4);
  float*    gcnacc = (float*)   carve((size_t)NNODES * HID * 4);

  const int  ETOT = NEDGES + NNODES;
  const long nXH  = (long)NNODES * HC;
  const long nNH  = (long)NNODES * HEADS;
  const long nNC  = (long)NNODES * HID;

  // 1) bf16 conversions; weights transposed so B fragments load contiguously
  cvt_bf16_k<<<(int)((nXH + 255) / 256), 256, 0, stream>>>(x, x_bf, nXH);
  transpose_cvt_k<<<(DIN * HC + 255) / 256, 256, 0, stream>>>(W_gat, WgatT, DIN, HC);
  transpose_cvt_k<<<(HC * HID + 255) / 256, 256, 0, stream>>>(W_gcn, WgcnT, HC, HID);

  // 2) GAT linear: h = x @ W_gat  (bf16 WMMA, f32 accumulate; 16x64 tile/wave)
  {
    int waves = (NNODES / 16) * (HC / 64);       // 12500 waves, one per block
    gemm_bf16_wmma_k<<<waves, 32, 0, stream>>>(x_bf, WgatT, hfeat, NNODES, HC, DIN);
  }

  // 3) attention coefficients
  attn_coeff_k<<<(int)((nNH + 255) / 256), 256, 0, stream>>>(hfeat, a_src, a_dst, asrc, adst, NNODES);

  // 4) init scatter buffers every call (harness doesn't reset workspace)
  fill_u32_k<<<(int)((nNH + 255) / 256), 256, 0, stream>>>(maxb, enc_host_minus_inf(), nNH);
  fill_f32_k<<<(int)((nNH + 255) / 256), 256, 0, stream>>>(sumb, 0.f, nNH);
  fill_f32_k<<<(NNODES + 255) / 256, 256, 0, stream>>>(deg, 0.f, (long)NNODES);
  fill_f32_k<<<(int)((nXH + 255) / 256), 256, 0, stream>>>(gatacc, 0.f, nXH);
  fill_f32_k<<<(int)((nNC + 255) / 256), 256, 0, stream>>>(gcnacc, 0.f, nNC);

  // 5) edge phases: segment max + degree -> exp-sum -> weighted message aggregation
  gat_edge_max_deg_k<<<(ETOT + 255) / 256, 256, 0, stream>>>(ei, NEDGES, NNODES, asrc, adst, maxb, deg);
  gat_edge_expsum_k<<<(ETOT + 255) / 256, 256, 0, stream>>>(ei, NEDGES, NNODES, asrc, adst, maxb, sumb);
  {
    long threads = (long)ETOT * HEADS * 32;      // one wave per (edge, head)
    gat_edge_message_k<<<(int)((threads + 255) / 256), 256, 0, stream>>>(
        ei, NEDGES, NNODES, hfeat, asrc, adst, maxb, sumb, gatacc);
  }

  // 6) x1 = relu(gat_out + b_gat) -> bf16 for next GEMM
  bias_relu_cvt_k<<<(int)((nXH + 255) / 256), 256, 0, stream>>>(gatacc, b_gat, x1_bf, nXH, HC);

  // 7) GCN linear: h2 = x1 @ W_gcn  (bf16 WMMA; N=64 -> one wave covers full N, A read once)
  {
    int waves = (NNODES / 16) * (HID / 64);      // 3125 waves
    gemm_bf16_wmma_k<<<waves, 32, 0, stream>>>(x1_bf, WgcnT, h2, NNODES, HID, HC);
  }

  // 8) GCN symmetric-normalized aggregation (one wave per edge)
  {
    long threads = (long)ETOT * 32;
    gcn_edge_k<<<(int)((threads + 255) / 256), 256, 0, stream>>>(ei, NEDGES, NNODES, h2, deg, gcnacc);
  }

  // 9) fused MLP head + sigmoid -> d_out [N]
  mlp_head_k<<<(NNODES + 255) / 256, 256, 0, stream>>>(gcnacc, b_gcn, W1, b1, W2, b2, out, NNODES);
}
